// DualAN_45466523796060
// MI455X (gfx1250) — compile-verified, matrix operand order checked by reading, not run on hardware
//
#include <hip/hip_runtime.h>
#include <hip/hip_bf16.h>

typedef __attribute__((ext_vector_type(16))) _Float16 v16h;
typedef __attribute__((ext_vector_type(8)))  float    v8f;

#define BB   32
#define SS   720
#define EE   862
#define MM   (BB*EE)        /* 27584 rows */
#define FF   361            /* rfft bins  */
#define NDFT 736            /* 722 padded to mult of 16 */
#define OUTC (4*EE)

// ---------------------------------------------------------------------------
// Generic FP16 WMMA GEMM: C[M,N] (+)= A[M,K] * B[K,N] (+bias)
// Block tile 128x128, 256 threads (8 waves, 4x2), wave tile 32x64
// (2x4 v_wmma_f32_16x16x32_f16). A staged [m][k]; B staged TRANSPOSED [n][k]
// so both fragment gathers are contiguous per lane -> ds_load_b128.
// flags bit0: accumulate into existing C.
// Requires K,N multiples of 8 (all call sites satisfy this).
// ---------------------------------------------------------------------------
__global__ __launch_bounds__(256) void gemm_wmma_f16(
    const _Float16* __restrict__ A, const _Float16* __restrict__ Bw,
    float* __restrict__ C, int M, int N, int K,
    const float* __restrict__ bias, int flags) {
  __shared__ _Float16 As[128][40];   // [m][k] 128x32, +8 pad (80B row)
  __shared__ _Float16 Bt[128][40];   // [n][k] 128x32 transposed, +8 pad

  const int tid  = threadIdx.x;
  const int lane = tid & 31;
  const int wid  = tid >> 5;
  const int wm   = (wid >> 1) * 32;        // wave M offset in block tile
  const int wn   = (wid & 1) * 64;         // wave N offset
  const int m0   = blockIdx.y * 128;
  const int n0   = blockIdx.x * 128;
  const int lm   = lane & 15;
  const int kb   = (lane >> 4) << 3;       // 0 or 8 (ISA 16-bit A/B layout)

  v8f acc[2][4];
  #pragma unroll
  for (int i = 0; i < 2; ++i)
    #pragma unroll
    for (int j = 0; j < 4; ++j) acc[i][j] = (v8f){};

  // A staging map: 2 threads per row, 16 halves each
  const int arow = tid >> 1, acol = (tid & 1) << 4;
  // B staging map: thread covers 2 consecutive k rows x 8 n columns
  const int bkp = (tid >> 4) << 1;         // k pair base: 0,2,..,30
  const int bns = (tid & 15) << 3;         // n segment: 0,8,..,120

  for (int k0 = 0; k0 < K; k0 += 32) {
    // speculative prefetch of next k-tile
    if (k0 + 32 < K) {
      int pm = (m0 + arow) < M ? (m0 + arow) : (M - 1);
      __builtin_prefetch((const void*)(A + (size_t)pm * K + (k0 + 32 + acol)), 0, 0);
      int pk = (k0 + 32 + bkp) < K ? (k0 + 32 + bkp) : (K - 1);
      int pn = (n0 + bns) < N ? (n0 + bns) : 0;
      __builtin_prefetch((const void*)(Bw + (size_t)pk * N + pn), 0, 0);
    }
    __syncthreads();
    { // ---- stage A tile (128x32) ----
      int gm = m0 + arow;
      if (gm < M && (k0 + acol + 16) <= K) {
        const _Float16* ap = A + (size_t)gm * K + k0 + acol;
        *(uint4*)&As[arow][acol]     = *(const uint4*)ap;
        *(uint4*)&As[arow][acol + 8] = *(const uint4*)(ap + 8);
      } else {   // branchless clamp+select tail
        int gmc = gm < M ? gm : (M - 1);
        #pragma unroll
        for (int j = 0; j < 16; ++j) {
          int gk  = k0 + acol + j;
          int gkc = gk < K ? gk : (K - 1);
          _Float16 t = A[(size_t)gmc * K + gkc];
          As[arow][acol + j] = (gm < M && gk < K) ? t : (_Float16)0.f;
        }
      }
    }
    { // ---- stage B tile transposed (32k x 128n -> Bt[n][k]) ----
      int gk0 = k0 + bkp, gk1 = gk0 + 1;
      int gn  = n0 + bns;
      if (gk1 < K && (gn + 8) <= N) {
        const _Float16* bp0 = Bw + (size_t)gk0 * N + gn;
        uint4 r0 = *(const uint4*)bp0;
        uint4 r1 = *(const uint4*)(bp0 + N);
        const _Float16* h0 = (const _Float16*)&r0;
        const _Float16* h1 = (const _Float16*)&r1;
        #pragma unroll
        for (int j = 0; j < 8; ++j) {
          union { _Float16 h[2]; unsigned u; } pk;
          pk.h[0] = h0[j]; pk.h[1] = h1[j];
          *(unsigned*)&Bt[bns + j][bkp] = pk.u;   // two k's per b32 store
        }
      } else {   // branchless clamp+select tail
        int k0c = gk0 < K ? gk0 : (K - 1);
        int k1c = gk1 < K ? gk1 : (K - 1);
        #pragma unroll
        for (int j = 0; j < 8; ++j) {
          int n  = gn + j;
          int nc = n < N ? n : (N - 1);
          _Float16 t0 = Bw[(size_t)k0c * N + nc];
          _Float16 t1 = Bw[(size_t)k1c * N + nc];
          union { _Float16 h[2]; unsigned u; } pk;
          pk.h[0] = (gk0 < K && n < N) ? t0 : (_Float16)0.f;
          pk.h[1] = (gk1 < K && n < N) ? t1 : (_Float16)0.f;
          *(unsigned*)&Bt[bns + j][bkp] = pk.u;
        }
      }
    }
    __syncthreads();

    v16h a0, a1, b[4];
    #pragma unroll
    for (int e = 0; e < 16; ++e) {
      const int kk = kb + (e & 7) + ((e >> 3) << 4);  // ISA table order: 8h+e, then 16+8h+(e-8)
      a0[e] = As[wm + lm][kk];
      a1[e] = As[wm + 16 + lm][kk];
      #pragma unroll
      for (int j = 0; j < 4; ++j)
        b[j][e] = Bt[wn + j * 16 + lm][kk];
    }
    #pragma unroll
    for (int j = 0; j < 4; ++j) {
      acc[0][j] = __builtin_amdgcn_wmma_f32_16x16x32_f16(false, a0, false, b[j], (short)0, acc[0][j], false, false);
      acc[1][j] = __builtin_amdgcn_wmma_f32_16x16x32_f16(false, a1, false, b[j], (short)0, acc[1][j], false, false);
    }
  }

  // ---- epilogue ----
  const int moff = (lane < 16) ? 0 : 8;
  #pragma unroll
  for (int i = 0; i < 2; ++i) {
    #pragma unroll
    for (int j = 0; j < 4; ++j) {
      int col = n0 + wn + j * 16 + lm;
      if (col < N) {
        float bv = bias ? bias[col] : 0.f;
        #pragma unroll
        for (int v = 0; v < 8; ++v) {
          int m = m0 + wm + i * 16 + moff + v;
          if (m < M) {
            size_t off = (size_t)m * N + col;
            float r = acc[i][j][v] + bv;
            if (flags & 1) r += C[off];
            C[off] = r;
          }
        }
      }
    }
  }
}

// ---------------------------------------------------------------------------
// DFT basis: Bd[t][f] = cos(2pi t f/720), Bd[t][361+f] = -sin(.), pad to 736.
// Integer phase reduction keeps accuracy for large t*f.
// ---------------------------------------------------------------------------
__global__ __launch_bounds__(256) void build_dft(_Float16* __restrict__ Bd) {
  int i = blockIdx.x * 256 + threadIdx.x;
  if (i >= SS * NDFT) return;
  int t = i / NDFT, c = i % NDFT;
  float v = 0.f;
  if (c < 2 * FF) {
    int f = (c < FF) ? c : (c - FF);
    int p = (int)(((long long)t * f) % SS);
    float th = 8.7266462e-3f * (float)p;   // 2*pi/720
    float s, cc;
    __sincosf(th, &s, &cc);
    v = (c < FF) ? cc : -s;
  }
  Bd[i] = (_Float16)v;
}

// x[B,S,E] -> xT[B*E, S]  (f32 and f16 copies), tiled via LDS
__global__ __launch_bounds__(256) void transpose_x(
    const float* __restrict__ x, float* __restrict__ xT,
    _Float16* __restrict__ xTh) {
  __shared__ float tile[32][33];
  int b = blockIdx.z, e0 = blockIdx.x * 32, s0 = blockIdx.y * 32;
  int tx = threadIdx.x & 31, ty = threadIdx.x >> 5;   // 32 x 8
  #pragma unroll
  for (int i = ty; i < 32; i += 8) {
    int s = s0 + i, e = e0 + tx;
    int sc = s < SS ? s : (SS - 1), ec = e < EE ? e : (EE - 1);
    float t = x[((size_t)b * SS + sc) * EE + ec];
    tile[i][tx] = (s < SS && e < EE) ? t : 0.f;
  }
  __syncthreads();
  #pragma unroll
  for (int i = ty; i < 32; i += 8) {
    int e = e0 + i, s = s0 + tx;
    if (e < EE && s < SS) {
      float v = tile[tx][i];
      size_t off = ((size_t)(b * EE + e)) * SS + s;
      xT[off] = v;
      xTh[off] = (_Float16)v;
    }
  }
}

__global__ __launch_bounds__(256) void cvt_f32_f16(
    const float* __restrict__ s, _Float16* __restrict__ d, long long n) {
  long long i = (long long)blockIdx.x * 256 + threadIdx.x;
  if (i < n) d[i] = (_Float16)s[i];
}

__global__ __launch_bounds__(256) void relu_cvt_f32_f16(
    const float* __restrict__ s, _Float16* __restrict__ d, long long n) {
  long long i = (long long)blockIdx.x * 256 + threadIdx.x;
  if (i < n) { float v = s[i]; d[i] = (_Float16)(v > 0.f ? v : 0.f); }
}

// ---------------------------------------------------------------------------
// Per-row top-20 |X_f| + 20-term inverse reconstruction.
// XF row layout: [re(0..360) | im(0..360) | pad..736)
// ---------------------------------------------------------------------------
__global__ __launch_bounds__(256) void topk_reconstruct(
    const float* __restrict__ XF, const float* __restrict__ xT,
    _Float16* __restrict__ xfilth, float* __restrict__ normin) {
  __shared__ float mag2[FF];
  __shared__ float redv[256];
  __shared__ int   redi[256];
  __shared__ float selRe[20], selIm[20];
  __shared__ int   selF[20];
  const int tid = threadIdx.x;
  const int r   = blockIdx.x;
  const size_t xb = (size_t)r * NDFT;

  for (int f = tid; f < FF; f += 256) {
    float re = XF[xb + f], im = XF[xb + FF + f];
    mag2[f] = re * re + im * im;
  }
  __syncthreads();

  for (int it = 0; it < 20; ++it) {
    float bv = -1.f; int bi = 0;
    for (int f = tid; f < FF; f += 256)
      if (mag2[f] > bv) { bv = mag2[f]; bi = f; }
    redv[tid] = bv; redi[tid] = bi;
    __syncthreads();
    for (int s = 128; s > 0; s >>= 1) {
      if (tid < s && redv[tid + s] > redv[tid]) {
        redv[tid] = redv[tid + s]; redi[tid] = redi[tid + s];
      }
      __syncthreads();
    }
    if (tid == 0) {
      int f = redi[0];
      selF[it] = f;
      float re = XF[xb + f], im = XF[xb + FF + f];
      float fac = ((f == 0) || (f == FF - 1)) ? (1.f / SS) : (2.f / SS);
      selRe[it] = fac * re;
      selIm[it] = -fac * im;
      mag2[f] = -2.f;
    }
    __syncthreads();
  }

  const size_t rb = (size_t)r * SS;
  for (int t = tid; t < SS; t += 256) {
    float acc = 0.f;
    #pragma unroll 5
    for (int k = 0; k < 20; ++k) {
      int p = (selF[k] * t) % SS;
      float th = 8.7266462e-3f * (float)p;
      float s, c;
      __sincosf(th, &s, &c);
      acc += selRe[k] * c + selIm[k] * s;
    }
    xfilth[rb + t] = (_Float16)acc;
    normin[rb + t] = xT[rb + t] - acc;
  }
}

// ---------------------------------------------------------------------------
// Sliding window-24 mean/std (replicate pad) per row; writes fp16 stat
// operands (mean rows 0..M-1, std rows M..2M-1) and the `norm` output block.
// ---------------------------------------------------------------------------
__global__ __launch_bounds__(256) void window_norm(
    const float* __restrict__ normin, _Float16* __restrict__ msb,
    float* __restrict__ out) {
  __shared__ float row[SS + 24];
  const int tid = threadIdx.x;
  const int r   = blockIdx.x;
  const size_t rb = (size_t)r * SS;
  for (int i = tid; i < SS + 24; i += 256) {
    int s = i - 12; s = s < 0 ? 0 : (s > SS - 1 ? SS - 1 : s);
    row[i] = normin[rb + s];
  }
  __syncthreads();
  const int b = r / EE, e = r % EE;
  for (int t = tid; t < SS; t += 256) {
    float s1 = 0.f, s2 = 0.f;
    #pragma unroll
    for (int i = 0; i < 24; ++i) { float v = row[t + i]; s1 += v; s2 += v * v; }
    float mean = s1 * (1.f / 24.f);
    float var  = fmaxf(s2 * (1.f / 24.f) - mean * mean, 0.f);
    float stdv = sqrtf(var + 1e-5f);
    msb[rb + t] = (_Float16)mean;
    msb[(size_t)MM * SS + rb + t] = (_Float16)stdv;
    out[((size_t)b * SS + t) * OUTC + e] = (row[t + 12] - mean) / stdv;
  }
}

// P[M,720] (row = b*E+e) -> out[b, t, col0 + e]
__global__ __launch_bounds__(256) void scatter_pred(
    const float* __restrict__ P, float* __restrict__ out, int col0) {
  size_t i = (size_t)blockIdx.x * 256 + threadIdx.x;
  if (i >= (size_t)MM * SS) return;
  int r = (int)(i / SS), t = (int)(i % SS);
  int b = r / EE, e = r % EE;
  out[((size_t)b * SS + t) * OUTC + col0 + e] = P[i];
}

// ---------------------------------------------------------------------------
static void launch_gemm(const _Float16* A, const _Float16* Bw, float* C,
                        int M, int N, int K, const float* bias, int flags,
                        hipStream_t st) {
  dim3 g((unsigned)((N + 127) / 128), (unsigned)((M + 127) / 128));
  gemm_wmma_f16<<<g, 256, 0, st>>>(A, Bw, C, M, N, K, bias, flags);
}

static inline size_t aln(size_t x) { return (x + 255) & ~(size_t)255; }

extern "C" void kernel_launch(void* const* d_in, const int* in_sizes, int n_in,
                              void* d_out, int out_size, void* d_ws, size_t ws_size,
                              hipStream_t stream) {
  (void)in_sizes; (void)n_in; (void)out_size; (void)ws_size;
  const float* x   = (const float*)d_in[0];
  const float* Wf1 = (const float*)d_in[1];  const float* bf1 = (const float*)d_in[2];
  const float* Wf2 = (const float*)d_in[3];  const float* bf2 = (const float*)d_in[4];
  const float* Wf3 = (const float*)d_in[5];  const float* bf3 = (const float*)d_in[6];
  const float* Wp1 = (const float*)d_in[7];  const float* bp1 = (const float*)d_in[8];
  const float* Wp2 = (const float*)d_in[9];  const float* bp2 = (const float*)d_in[10];
  const float* Wp3 = (const float*)d_in[11]; const float* bp3 = (const float*)d_in[12];
  float* out = (float*)d_out;

  // ---- workspace carve ----
  char* p = (char*)d_ws; size_t off = 0;
  auto take = [&](size_t bytes) { char* r = p + off; off = aln(off + bytes); return r; };

  float*    xT      = (float*)   take((size_t)MM * SS * 4);
  _Float16* xTh     = (_Float16*)take((size_t)MM * SS * 2);
  _Float16* Bdft    = (_Float16*)take((size_t)SS * NDFT * 2);
  _Float16* Wf1h    = (_Float16*)take((size_t)SS * 64 * 2);
  _Float16* Wf2h    = (_Float16*)take((size_t)(64 + SS) * 128 * 2);
  _Float16* Wf3h    = (_Float16*)take((size_t)128 * SS * 2);
  _Float16* Wp1h    = (_Float16*)take((size_t)SS * 256 * 2);
  _Float16* Wp2h    = (_Float16*)take((size_t)(256 + SS) * 512 * 2);
  _Float16* Wp3h    = (_Float16*)take((size_t)512 * SS * 2);
  float*    XF      = (float*)   take((size_t)MM * NDFT * 4);   // reused as Pmain
  _Float16* xfilth  = (_Float16*)take((size_t)MM * SS * 2);
  float*    normin  = (float*)   take((size_t)MM * SS * 4);
  _Float16* msb     = (_Float16*)take((size_t)2 * MM * SS * 2); // mean | std stacked
  float*    Hf      = (float*)   take((size_t)MM * 64 * 4);
  _Float16* Hfh     = (_Float16*)take((size_t)MM * 64 * 2);
  float*    H2f     = (float*)   take((size_t)MM * 128 * 4);
  _Float16* H2fh    = (_Float16*)take((size_t)MM * 128 * 2);
  float*    Hms     = (float*)   take((size_t)2 * MM * 256 * 4);
  _Float16* Hmsh    = (_Float16*)take((size_t)2 * MM * 256 * 2);
  float*    H2ms    = (float*)   take((size_t)2 * MM * 512 * 4);
  _Float16* H2msh   = (_Float16*)take((size_t)2 * MM * 512 * 2);
  float*    Pms     = (float*)   take((size_t)2 * MM * SS * 4);
  float*    Pmain   = XF;   // XF fully consumed before Pmain is produced

  auto nblk = [](long long n) { return (unsigned)((n + 255) / 256); };

  // ---- stage 0: operand prep ----
  build_dft<<<nblk((long long)SS * NDFT), 256, 0, stream>>>(Bdft);
  {
    dim3 g((EE + 31) / 32, (SS + 31) / 32, BB);
    transpose_x<<<g, 256, 0, stream>>>(x, xT, xTh);
  }
  cvt_f32_f16<<<nblk((long long)SS * 64), 256, 0, stream>>>(Wf1, Wf1h, (long long)SS * 64);
  cvt_f32_f16<<<nblk((long long)(64 + SS) * 128), 256, 0, stream>>>(Wf2, Wf2h, (long long)(64 + SS) * 128);
  cvt_f32_f16<<<nblk((long long)128 * SS), 256, 0, stream>>>(Wf3, Wf3h, (long long)128 * SS);
  cvt_f32_f16<<<nblk((long long)SS * 256), 256, 0, stream>>>(Wp1, Wp1h, (long long)SS * 256);
  cvt_f32_f16<<<nblk((long long)(256 + SS) * 512), 256, 0, stream>>>(Wp2, Wp2h, (long long)(256 + SS) * 512);
  cvt_f32_f16<<<nblk((long long)512 * SS), 256, 0, stream>>>(Wp3, Wp3h, (long long)512 * SS);

  // ---- stage 1: rfft as GEMM, top-k filter, window stats ----
  launch_gemm(xTh, Bdft, XF, MM, NDFT, SS, nullptr, 0, stream);
  topk_reconstruct<<<MM, 256, 0, stream>>>(XF, xT, xfilth, normin);
  window_norm<<<MM, 256, 0, stream>>>(normin, msb, out);   // writes norm block

  // ---- stage 2: freq MLP -> pred_main ----
  launch_gemm(xfilth, Wf1h, Hf, MM, 64, SS, bf1, 0, stream);
  relu_cvt_f32_f16<<<nblk((long long)MM * 64), 256, 0, stream>>>(Hf, Hfh, (long long)MM * 64);
  launch_gemm(Hfh, Wf2h, H2f, MM, 128, 64, nullptr, 0, stream);                // h part
  launch_gemm(xTh, Wf2h + (size_t)64 * 128, H2f, MM, 128, SS, bf2, 1, stream); // x part (acc)
  relu_cvt_f32_f16<<<nblk((long long)MM * 128), 256, 0, stream>>>(H2f, H2fh, (long long)MM * 128);
  launch_gemm(H2fh, Wf3h, Pmain, MM, SS, 128, bf3, 0, stream);
  scatter_pred<<<nblk((long long)MM * SS), 256, 0, stream>>>(Pmain, out, EE);

  // ---- stage 3: pred MLP on stacked [mean; std] -> pred_means / pred_stds ----
  launch_gemm(msb, Wp1h, Hms, 2 * MM, 256, SS, bp1, 0, stream);
  relu_cvt_f32_f16<<<nblk((long long)2 * MM * 256), 256, 0, stream>>>(Hms, Hmsh, (long long)2 * MM * 256);
  launch_gemm(Hmsh, Wp2h, H2ms, 2 * MM, 512, 256, nullptr, 0, stream);           // h part
  launch_gemm(xTh, Wp2h + (size_t)256 * 512, H2ms, MM, 512, SS, bp2, 1, stream);              // x (mean half)
  launch_gemm(xTh, Wp2h + (size_t)256 * 512, H2ms + (size_t)MM * 512, MM, 512, SS, bp2, 1, stream); // x (std half)
  relu_cvt_f32_f16<<<nblk((long long)2 * MM * 512), 256, 0, stream>>>(H2ms, H2msh, (long long)2 * MM * 512);
  launch_gemm(H2msh, Wp3h, Pms, 2 * MM, SS, 512, bp3, 0, stream);
  scatter_pred<<<nblk((long long)MM * SS), 256, 0, stream>>>(Pms, out, 2 * EE);
  scatter_pred<<<nblk((long long)MM * SS), 256, 0, stream>>>(Pms + (size_t)MM * SS, out, 3 * EE);
}
